// HybridGraphBlockK8_3032246911437
// MI455X (gfx1250) — matrix operand order repaired and checked by reading, not compiled, not run on hardware
//
#include <hip/hip_runtime.h>
#include <hip/hip_bf16.h>
#include <math.h>

// ---------------------------------------------------------------------------
// HybridGraphBlockK8 on MI455X (gfx1250, wave32, WMMA + Tensor Data Mover).
//
// Roofline: ~270 GFLOP of dense GEMM vs ~1 GB of activation traffic at
// 23.3 TB/s -> compute and memory are balanced only if the GEMMs run on the
// WMMA pipes in f16 (f32 accumulate). All six GEMMs use
// V_WMMA_F32_16X16X32_F16 with LDS-staged tiles; the A tile is staged by the
// TDM (tensor_load_to_lds) with descriptor padding giving the 40-half LDS row
// stride, overlapped with VALU-side B-tile transpose staging.
// ---------------------------------------------------------------------------

typedef _Float16 v16h __attribute__((ext_vector_type(16)));
typedef float    v8f  __attribute__((ext_vector_type(8)));
typedef unsigned int u32x4 __attribute__((ext_vector_type(4)));
typedef int      i32x4 __attribute__((ext_vector_type(4)));
typedef int      i32x8 __attribute__((ext_vector_type(8)));

#define D_MODEL 512
#define NHEAD   8
#define HD      64
#define HIDDEN  1536

#if defined(__gfx1250__) && __has_builtin(__builtin_amdgcn_tensor_load_to_lds)
#define HAVE_TDM 1
#else
#define HAVE_TDM 0
#endif

__device__ __forceinline__ void wait_tensorcnt0() {
#if __has_builtin(__builtin_amdgcn_s_wait_tensorcnt)
  __builtin_amdgcn_s_wait_tensorcnt(0);
#else
  asm volatile("s_wait_tensorcnt 0x0" ::: "memory");
#endif
}

// ---------------------------------------------------------------------------
// f32 -> f16 elementwise (weights; ~10 MB total, negligible cost)
// ---------------------------------------------------------------------------
__global__ void __launch_bounds__(256) f32_to_f16_kernel(const float* __restrict__ in,
                                                         _Float16* __restrict__ out,
                                                         int n) {
  int i = blockIdx.x * 256 + threadIdx.x;
  if (i < n) out[i] = (_Float16)in[i];
}

// ---------------------------------------------------------------------------
// LayerNorm over 512 cols. One wave (32 lanes) per row, 8 rows per block.
// Lane l owns columns {l, l+32, ..., l+480}; shuffle reduction (wave32).
// ---------------------------------------------------------------------------
__global__ void __launch_bounds__(256) ln_f16_kernel(const float* __restrict__ x,
                                                     const float* __restrict__ g,
                                                     const float* __restrict__ b,
                                                     _Float16* __restrict__ out,
                                                     int N) {
  const int wave = threadIdx.x >> 5;
  const int lane = threadIdx.x & 31;
  const int row  = blockIdx.x * 8 + wave;
  if (row >= N) return;
  const float* xr = x + (size_t)row * D_MODEL;

  float vals[16];
  float s = 0.f, s2 = 0.f;
#pragma unroll
  for (int i = 0; i < 16; ++i) {
    float v = xr[lane + i * 32];
    vals[i] = v;
    s += v;
    s2 += v * v;
  }
#pragma unroll
  for (int o = 16; o >= 1; o >>= 1) {
    s  += __shfl_xor(s, o, 32);
    s2 += __shfl_xor(s2, o, 32);
  }
  const float mean = s * (1.0f / D_MODEL);
  const float var  = s2 * (1.0f / D_MODEL) - mean * mean;
  const float rstd = rsqrtf(var + 1e-5f);

  _Float16* orow = out + (size_t)row * D_MODEL;
#pragma unroll
  for (int i = 0; i < 16; ++i) {
    int c = lane + i * 32;
    orow[c] = (_Float16)((vals[i] - mean) * rstd * g[c] + b[c]);
  }
}

// ---------------------------------------------------------------------------
// WMMA GEMM: C[M,Ncols] = A[M,K](f16) @ B[K,Ncols](f16) + bias (+resid)(+gelu)
//
// Block: 256 threads = 8 waves. Block tile 128x64, K-tile 32.
// Wave w owns output rows [w*16, w*16+16) x all 64 cols: 4 v8f accumulators
// driven by V_WMMA_F32_16X16X32_F16.
//
// A tile: staged by TDM (tensor_load_to_lds), one descriptor per K-step,
//   issued by wave 0. D# padding (interval = 16 DW / 64 B row, amount = 4 DW)
//   realizes the bank-conflict-avoiding 40-half LDS row stride; tensor_dim1 =
//   rows-remaining gives hardware zero-fill for the ragged last block.
// B tile: stored TRANSPOSED [col][K] so the B fragment gather (K-contiguous
//   per lane) is a contiguous ds_load_b128 pair; staged by all 256 threads
//   concurrently with the TDM transfer.
//
// Fragment layouts follow CDNA5 ISA 7.12.2 (16-bit A 16x32, B 32x16):
//   A: lane l (row = l&15, g = l>>4): halves 0..7 -> K = g*8 + i,
//      halves 8..15 -> K = 16 + g*8 + (i-8)
//   B: lane l (col = l&15, g = l>>4): same K mapping as A
//   C: VGPR r: row = (l>>4)*8 + r, col = l&15
// ---------------------------------------------------------------------------
template <bool GELU, bool RESID, bool OUT16>
__global__ void __launch_bounds__(256) gemm_wmma_kernel(
    const _Float16* __restrict__ A, const _Float16* __restrict__ B,
    const float* __restrict__ bias, const float* __restrict__ resid,
    float* __restrict__ Cf, _Float16* __restrict__ Ch,
    int M, int K, int Ncols) {
  constexpr int BM = 128, BN = 64, BK = 32;
  constexpr int AST = BK + 8;  // 40 halves (80 B): 64 B data + 16 B pad
  constexpr int BST = BK + 8;  // B^T stride
  __shared__ _Float16 As[BM * AST];
  __shared__ _Float16 Bs[BN * BST];

  const int tid  = threadIdx.x;
  const int wave = tid >> 5;
  const int lane = tid & 31;
  const int row0 = blockIdx.x * BM;
  const int col0 = blockIdx.y * BN;
  const int lrow = lane & 15;
  const int lgrp = lane >> 4;

  v8f acc[4] = {};

#if HAVE_TDM
  const int rows_avail = (M - row0) < BM ? (M - row0) : BM;
  const unsigned lds_base = (unsigned)(unsigned long long)(void*)&As[0];
#endif

  for (int kk = 0; kk < K; kk += BK) {
#if HAVE_TDM
    // ---- A tile via Tensor Data Mover (wave 0 issues; EXEC-independent DMA)
    if (wave == 0) {
      const unsigned long long ga =
          (unsigned long long)(const void*)(A + (size_t)row0 * K + kk);
      u32x4 g0;
      g0[0] = 1u;                                   // count=1, user mode
      g0[1] = lds_base;                             // lds_addr (bytes)
      g0[2] = (unsigned)(ga & 0xFFFFFFFFu);         // global_addr[31:0]
      g0[3] = (unsigned)((ga >> 32) & 0x01FFFFFFu)  // global_addr[56:32]
              | (2u << 30);                         // type = 2 ("image")
      i32x8 g1;
      g1[0] = (1 << 16)     // data_size = 1 -> 2 bytes
            | (1 << 20)     // pad_enable
            | (3 << 22)     // pad_interval: 16 DWORDs (64 B = one tile row)
            | (3 << 25);    // pad_amount:   4 DWORDs (16 B -> 80 B stride)
      g1[1] = (int)(BK << 16);                      // tensor_dim0 = 32
      g1[2] = (int)((rows_avail & 0xFFFF) << 16);   // tensor_dim1 (OOB->0)
      g1[3] = (int)(BK << 16);                      // tile_dim0 = 32
      g1[4] = BM;                                   // tile_dim1 = 128
      g1[5] = K;                                    // tensor_dim0_stride
      g1[6] = 0;
      g1[7] = 0;
      i32x4 gz4; gz4[0] = 0; gz4[1] = 0; gz4[2] = 0; gz4[3] = 0;
      i32x8 gz8;
#pragma unroll
      for (int i = 0; i < 8; ++i) gz8[i] = 0;
      // 6-arg form (clang-23 / therock headers): groups 0..3 + extra + cpol
      __builtin_amdgcn_tensor_load_to_lds(g0, g1, gz4, gz4, gz8, 0);
    }
#else
    // ---- fallback: stage A tile via VGPRs, 2 threads/row, uint4 loads
    {
      const int arow = tid >> 1;
      const int aoff = (tid & 1) * 16;
      uint4 av = make_uint4(0u, 0u, 0u, 0u);
      uint4 av2 = make_uint4(0u, 0u, 0u, 0u);
      if (row0 + arow < M) {
        av  = *(const uint4*)(A + (size_t)(row0 + arow) * K + kk + aoff);
        av2 = *(const uint4*)(A + (size_t)(row0 + arow) * K + kk + aoff + 8);
      }
      *(uint4*)&As[arow * AST + aoff] = av;
      *(uint4*)&As[arow * AST + aoff + 8] = av2;
    }
#endif
    // ---- stage B tile transposed: 32 K-rows x 64 cols -> Bs[col][K]
    //      (overlaps with the in-flight TDM transfer)
    {
      const int krel   = tid >> 3;        // 0..31
      const int colrel = (tid & 7) * 8;   // 0..56
      uint4 bv = *(const uint4*)(B + (size_t)(kk + krel) * Ncols + col0 + colrel);
      const _Float16* bh = (const _Float16*)&bv;
#pragma unroll
      for (int j = 0; j < 8; ++j) Bs[(colrel + j) * BST + krel] = bh[j];
    }
#if HAVE_TDM
    if (wave == 0) wait_tensorcnt0();   // drain TENSORcnt before the barrier
#endif
    __syncthreads();

    // ---- gather A fragment for this wave's 16-row strip
    v16h af;
    {
      const _Float16* ap = &As[(wave * 16 + lrow) * AST];
#pragma unroll
      for (int i = 0; i < 8; ++i) af[i] = ap[lgrp * 8 + i];
#pragma unroll
      for (int i = 0; i < 8; ++i) af[8 + i] = ap[16 + lgrp * 8 + i];
    }
    // ---- gather ALL four B fragments first, then issue WMMAs back-to-back
    v16h bf[4];
#pragma unroll
    for (int ct = 0; ct < 4; ++ct) {
      const _Float16* bp = &Bs[(ct * 16 + lrow) * BST];
#pragma unroll
      for (int i = 0; i < 16; ++i) bf[ct][i] = bp[(i < 8) ? (lgrp * 8 + i) : (16 + lgrp * 8 + i - 8)];
    }
#pragma unroll
    for (int ct = 0; ct < 4; ++ct) {
      acc[ct] = __builtin_amdgcn_wmma_f32_16x16x32_f16(
          false, af, false, bf[ct], (short)0, acc[ct], false, false);
    }
    __syncthreads();
  }

  // ---- epilogue
#pragma unroll
  for (int ct = 0; ct < 4; ++ct) {
#pragma unroll
    for (int r = 0; r < 8; ++r) {
      const int row = row0 + wave * 16 + lgrp * 8 + r;
      const int col = col0 + ct * 16 + lrow;
      if (row < M) {
        float val = acc[ct][r] + bias[col];
        if (RESID) val += resid[(size_t)row * Ncols + col];
        if (GELU)  val = 0.5f * val * (1.0f + erff(val * 0.70710678118654752f));
        if (OUT16) Ch[(size_t)row * Ncols + col] = (_Float16)val;
        else       Cf[(size_t)row * Ncols + col] = val;
      }
    }
  }
}

// ---------------------------------------------------------------------------
// k=8 (+self) attention. One block per node, one wave per head (H=8 waves).
// Lane l owns head-dim elements {2l, 2l+1}. Scores via shuffle reduction,
// softmax over 9 keys in registers. K/V rows prefetched (global_prefetch_b8).
// ---------------------------------------------------------------------------
__global__ void __launch_bounds__(256) attn_k8_kernel(
    const _Float16* __restrict__ q, const _Float16* __restrict__ k,
    const _Float16* __restrict__ v, const int* __restrict__ idx,
    _Float16* __restrict__ out, int N) {
  const int n    = blockIdx.x;
  const int head = threadIdx.x >> 5;
  const int lane = threadIdx.x & 31;
  if (n >= N) return;

  const size_t hoff = (size_t)head * HD + lane * 2;
  const _Float16* qp = q + (size_t)n * D_MODEL + hoff;
  const float q0 = (float)qp[0], q1 = (float)qp[1];

  int nbr[9];
  nbr[0] = n;
#pragma unroll
  for (int j = 1; j < 9; ++j) nbr[j] = idx[(size_t)n * 8 + (j - 1)];
#pragma unroll
  for (int j = 0; j < 9; ++j) {
    __builtin_prefetch(k + (size_t)nbr[j] * D_MODEL + hoff, 0, 0);
    __builtin_prefetch(v + (size_t)nbr[j] * D_MODEL + hoff, 0, 0);
  }

  float sc[9];
#pragma unroll
  for (int j = 0; j < 9; ++j) {
    const _Float16* kp = k + (size_t)nbr[j] * D_MODEL + hoff;
    float p = q0 * (float)kp[0] + q1 * (float)kp[1];
#pragma unroll
    for (int o = 16; o >= 1; o >>= 1) p += __shfl_xor(p, o, 32);
    sc[j] = p * 0.125f;  // 1/sqrt(64)
  }
  float m = sc[0];
#pragma unroll
  for (int j = 1; j < 9; ++j) m = fmaxf(m, sc[j]);
  float den = 0.f;
#pragma unroll
  for (int j = 0; j < 9; ++j) { sc[j] = __expf(sc[j] - m); den += sc[j]; }
  const float inv = 1.0f / den;

  float o0 = 0.f, o1 = 0.f;
#pragma unroll
  for (int j = 0; j < 9; ++j) {
    const _Float16* vp = v + (size_t)nbr[j] * D_MODEL + hoff;
    const float w = sc[j] * inv;
    o0 += w * (float)vp[0];
    o1 += w * (float)vp[1];
  }
  _Float16* op = out + (size_t)n * D_MODEL + hoff;
  op[0] = (_Float16)o0;
  op[1] = (_Float16)o1;
}

// ---------------------------------------------------------------------------
// Host-side orchestration
// ---------------------------------------------------------------------------
extern "C" void kernel_launch(void* const* d_in, const int* in_sizes, int n_in,
                              void* d_out, int out_size, void* d_ws, size_t ws_size,
                              hipStream_t stream) {
  const float* x     = (const float*)d_in[0];
  const int*   idx   = (const int*)d_in[1];
  const float* ln1_g = (const float*)d_in[2];
  const float* ln1_b = (const float*)d_in[3];
  const float* Wq    = (const float*)d_in[4];
  const float* bq    = (const float*)d_in[5];
  const float* Wk    = (const float*)d_in[6];
  const float* bk    = (const float*)d_in[7];
  const float* Wv    = (const float*)d_in[8];
  const float* bv    = (const float*)d_in[9];
  const float* Wo    = (const float*)d_in[10];
  const float* bo    = (const float*)d_in[11];
  const float* ln2_g = (const float*)d_in[12];
  const float* ln2_b = (const float*)d_in[13];
  const float* W1    = (const float*)d_in[14];
  const float* b1    = (const float*)d_in[15];
  const float* W2    = (const float*)d_in[16];
  const float* b2    = (const float*)d_in[17];
  float* out = (float*)d_out;

  const int N = in_sizes[0] / D_MODEL;

  // ---- workspace carve-up (256B aligned)
  char* ws = (char*)d_ws;
  size_t off = 0;
  auto carve = [&](size_t bytes) -> char* {
    char* p = ws + off;
    off = (off + bytes + 255) & ~(size_t)255;
    return p;
  };
  const size_t ND = (size_t)N * D_MODEL;
  _Float16* xn16   = (_Float16*)carve(ND * 2);
  _Float16* q16    = (_Float16*)carve(ND * 2);
  _Float16* k16    = (_Float16*)carve(ND * 2);
  _Float16* v16    = (_Float16*)carve(ND * 2);
  _Float16* attn16 = (_Float16*)carve(ND * 2);
  float*    x1     = (float*)carve(ND * 4);
  _Float16* xn2_16 = (_Float16*)carve(ND * 2);
  _Float16* h16    = (_Float16*)carve((size_t)N * HIDDEN * 2);
  _Float16* Wq16   = (_Float16*)carve((size_t)D_MODEL * D_MODEL * 2);
  _Float16* Wk16   = (_Float16*)carve((size_t)D_MODEL * D_MODEL * 2);
  _Float16* Wv16   = (_Float16*)carve((size_t)D_MODEL * D_MODEL * 2);
  _Float16* Wo16   = (_Float16*)carve((size_t)D_MODEL * D_MODEL * 2);
  _Float16* W116   = (_Float16*)carve((size_t)D_MODEL * HIDDEN * 2);
  _Float16* W216   = (_Float16*)carve((size_t)HIDDEN * D_MODEL * 2);
  (void)ws_size; (void)n_in; (void)out_size;

  // ---- convert weights to f16 (one pass, tiny)
  auto cvt = [&](const float* src, _Float16* dst, int n) {
    f32_to_f16_kernel<<<(n + 255) / 256, 256, 0, stream>>>(src, dst, n);
  };
  cvt(Wq, Wq16, D_MODEL * D_MODEL);
  cvt(Wk, Wk16, D_MODEL * D_MODEL);
  cvt(Wv, Wv16, D_MODEL * D_MODEL);
  cvt(Wo, Wo16, D_MODEL * D_MODEL);
  cvt(W1, W116, D_MODEL * HIDDEN);
  cvt(W2, W216, HIDDEN * D_MODEL);

  const int mtiles = (N + 127) / 128;

  // ---- LN1
  ln_f16_kernel<<<(N + 7) / 8, 256, 0, stream>>>(x, ln1_g, ln1_b, xn16, N);

  // ---- Q/K/V projections (WMMA f16)
  dim3 gQKV(mtiles, D_MODEL / 64);
  gemm_wmma_kernel<false, false, true><<<gQKV, 256, 0, stream>>>(
      xn16, Wq16, bq, nullptr, nullptr, q16, N, D_MODEL, D_MODEL);
  gemm_wmma_kernel<false, false, true><<<gQKV, 256, 0, stream>>>(
      xn16, Wk16, bk, nullptr, nullptr, k16, N, D_MODEL, D_MODEL);
  gemm_wmma_kernel<false, false, true><<<gQKV, 256, 0, stream>>>(
      xn16, Wv16, bv, nullptr, nullptr, v16, N, D_MODEL, D_MODEL);

  // ---- k=8 (+self) attention, one block per node, wave per head
  attn_k8_kernel<<<N, 256, 0, stream>>>(q16, k16, v16, idx, attn16, N);

  // ---- Wo projection + residual -> x1 (f32)
  gemm_wmma_kernel<false, true, false><<<gQKV, 256, 0, stream>>>(
      attn16, Wo16, bo, x, x1, nullptr, N, D_MODEL, D_MODEL);

  // ---- LN2
  ln_f16_kernel<<<(N + 7) / 8, 256, 0, stream>>>(x1, ln2_g, ln2_b, xn2_16, N);

  // ---- MLP up-projection + exact GELU -> h (f16)
  dim3 gUp(mtiles, HIDDEN / 64);
  gemm_wmma_kernel<true, false, true><<<gUp, 256, 0, stream>>>(
      xn2_16, W116, b1, nullptr, nullptr, h16, N, D_MODEL, HIDDEN);

  // ---- MLP down-projection + residual -> out (f32)
  dim3 gDn(mtiles, D_MODEL / 64);
  gemm_wmma_kernel<false, true, false><<<gDn, 256, 0, stream>>>(
      h16, W216, b2, x1, out, nullptr, N, HIDDEN, D_MODEL);
}
